// GraphAttentionPool_37563783971460
// MI455X (gfx1250) — matrix operand order repaired and checked by reading, not compiled, not run on hardware
//
#include <hip/hip_runtime.h>
#include <hip/hip_bf16.h>

typedef __attribute__((ext_vector_type(16))) __bf16 v16bf;
typedef __attribute__((ext_vector_type(8)))  __bf16 v8bf;
typedef __attribute__((ext_vector_type(8)))  float  v8f;

static __device__ inline v16bf cat8(v8bf lo, v8bf hi) {
    union { v16bf v; v8bf h[2]; } u;
    u.h[0] = lo; u.h[1] = hi;
    return u.v;
}

static __device__ inline float leaky(float v) { return v >= 0.0f ? v : 0.2f * v; }

// order-preserving float <-> uint for atomicMax over signed floats
static __device__ inline unsigned fenc(float x) {
    unsigned b = __float_as_uint(x);
    return (b & 0x80000000u) ? ~b : (b | 0x80000000u);
}
static __device__ inline float fdec(unsigned u) {
    unsigned b = (u & 0x80000000u) ? (u & 0x7fffffffu) : ~u;
    return __uint_as_float(b);
}

#define LD8(p)  (*reinterpret_cast<const v8bf*>(p))
#define WMMA_BF16(a, b, c) __builtin_amdgcn_wmma_f32_16x16x32_bf16( \
    false, (a), false, (b), (short)0, (c), false, false)

// ---------------------------------------------------------------------------
// bf16 WMMA GEMM: C[M,ldc] = A[M,K] (row-major bf16) x Bt[ldc,K]^T + bias.
// One wave computes a 16x64 tile. K is compile-time so B-column offsets
// (j*16*K) are VMEM immediates from a single base pointer. The K-loop is kept
// as a loop (unroll 1) with explicit register ping-pong: next chunk's 10 b128
// loads are issued before the current chunk's 4 WMMAs, giving the scheduler a
// fixed >=1-chunk load/compute overlap per wave. Requires K % 64 == 0.
// ---------------------------------------------------------------------------
template<int K>
__global__ __launch_bounds__(32)
void gemm_bf16_wmma(const __bf16* __restrict__ A,  long sA,
                    const __bf16* __restrict__ Bt, long sB,
                    float* __restrict__ C,         long sC,
                    const float* __restrict__ bias,
                    int ldc)
{
    const int lane = threadIdx.x;
    const int m0 = blockIdx.x << 4;
    const int n0 = blockIdx.y << 6;
    const int s  = blockIdx.z;
    A  += (long)s * sA;
    Bt += (long)s * sB;
    C  += (long)s * sC;

    const int lr   = lane & 15;
    const int hi16 = lane >> 4;

    // lane-resolved base pointers (A: +16 K for upper half-wave, B: +16 K)
    const __bf16* ap = A  + (long)(m0 + lr) * K + hi16 * 16;
    const __bf16* bp = Bt + (long)(n0 + lr) * K + hi16 * 16;

    v8f acc0 = {}, acc1 = {}, acc2 = {}, acc3 = {};

    // prologue: K-chunk 0 -> buffer set 0
    v16bf a0  = cat8(LD8(ap),              LD8(ap + 8));
    v16bf b00 = cat8(LD8(bp + 0 * 16 * K), LD8(bp + 0 * 16 * K + 8));
    v16bf b01 = cat8(LD8(bp + 1 * 16 * K), LD8(bp + 1 * 16 * K + 8));
    v16bf b02 = cat8(LD8(bp + 2 * 16 * K), LD8(bp + 2 * 16 * K + 8));
    v16bf b03 = cat8(LD8(bp + 3 * 16 * K), LD8(bp + 3 * 16 * K + 8));

    // steady state: each iteration consumes two 32-wide K chunks
#pragma unroll 1
    for (int it = (K >> 6) - 1; it > 0; --it) {
        // odd chunk -> buffer set 1 (loads overlap the buffer-0 WMMAs below)
        v16bf a1  = cat8(LD8(ap + 32),              LD8(ap + 40));
        v16bf b10 = cat8(LD8(bp + 0 * 16 * K + 32), LD8(bp + 0 * 16 * K + 40));
        v16bf b11 = cat8(LD8(bp + 1 * 16 * K + 32), LD8(bp + 1 * 16 * K + 40));
        v16bf b12 = cat8(LD8(bp + 2 * 16 * K + 32), LD8(bp + 2 * 16 * K + 40));
        v16bf b13 = cat8(LD8(bp + 3 * 16 * K + 32), LD8(bp + 3 * 16 * K + 40));
        acc0 = WMMA_BF16(a0, b00, acc0);
        acc1 = WMMA_BF16(a0, b01, acc1);
        acc2 = WMMA_BF16(a0, b02, acc2);
        acc3 = WMMA_BF16(a0, b03, acc3);
        // next even chunk reloads buffer set 0 in place (no copies)
        a0  = cat8(LD8(ap + 64),              LD8(ap + 72));
        b00 = cat8(LD8(bp + 0 * 16 * K + 64), LD8(bp + 0 * 16 * K + 72));
        b01 = cat8(LD8(bp + 1 * 16 * K + 64), LD8(bp + 1 * 16 * K + 72));
        b02 = cat8(LD8(bp + 2 * 16 * K + 64), LD8(bp + 2 * 16 * K + 72));
        b03 = cat8(LD8(bp + 3 * 16 * K + 64), LD8(bp + 3 * 16 * K + 72));
        acc0 = WMMA_BF16(a1, b10, acc0);
        acc1 = WMMA_BF16(a1, b11, acc1);
        acc2 = WMMA_BF16(a1, b12, acc2);
        acc3 = WMMA_BF16(a1, b13, acc3);
        ap += 64; bp += 64;
    }

    // epilogue: final odd chunk, then drain both buffers
    {
        v16bf a1  = cat8(LD8(ap + 32),              LD8(ap + 40));
        v16bf b10 = cat8(LD8(bp + 0 * 16 * K + 32), LD8(bp + 0 * 16 * K + 40));
        v16bf b11 = cat8(LD8(bp + 1 * 16 * K + 32), LD8(bp + 1 * 16 * K + 40));
        v16bf b12 = cat8(LD8(bp + 2 * 16 * K + 32), LD8(bp + 2 * 16 * K + 40));
        v16bf b13 = cat8(LD8(bp + 3 * 16 * K + 32), LD8(bp + 3 * 16 * K + 40));
        acc0 = WMMA_BF16(a0, b00, acc0);
        acc1 = WMMA_BF16(a0, b01, acc1);
        acc2 = WMMA_BF16(a0, b02, acc2);
        acc3 = WMMA_BF16(a0, b03, acc3);
        acc0 = WMMA_BF16(a1, b10, acc0);
        acc1 = WMMA_BF16(a1, b11, acc1);
        acc2 = WMMA_BF16(a1, b12, acc2);
        acc3 = WMMA_BF16(a1, b13, acc3);
    }

    v8f accs[4] = {acc0, acc1, acc2, acc3};
#pragma unroll
    for (int j = 0; j < 4; ++j) {
        const int col = n0 + j * 16 + lr;
        const float badd = bias ? bias[col] : 0.0f;
#pragma unroll
        for (int r = 0; r < 8; ++r) {
            const int row = m0 + r + hi16 * 8;
            C[(long)row * ldc + col] = accs[j][r] + badd;
        }
    }
}

// ---------------------------------------------------------------------------
// Helpers: zero / f32->bf16 / transposing f32->bf16 ([S][R][C] -> [S][C][R])
// ---------------------------------------------------------------------------
__global__ void zero_kernel(float* __restrict__ p, long total) {
    long i = (long)blockIdx.x * 256 + threadIdx.x;
    if (i < total) p[i] = 0.0f;
}

__global__ void f2bf_kernel(const float* __restrict__ in, __bf16* __restrict__ out, long total) {
    long i = (long)blockIdx.x * 256 + threadIdx.x;
    if (i < total) out[i] = (__bf16)in[i];
}

__global__ void transpose_f2bf(const float* __restrict__ in, __bf16* __restrict__ out,
                               int R, int Ccols, long total) {
    long idx = (long)blockIdx.x * 256 + threadIdx.x;   // idx = (s*C + c)*R + r
    if (idx >= total) return;
    long sc = idx / R;
    int  r  = (int)(idx % R);
    long s  = sc / Ccols;
    int  c  = (int)(sc % Ccols);
    out[idx] = (__bf16)in[(s * R + r) * (long)Ccols + c];
}

// ---------------------------------------------------------------------------
// Per-(scale,node,head) attention dot products: one wave, 4 floats/lane, F=128
// ---------------------------------------------------------------------------
__global__ __launch_bounds__(256)
void sdots_kernel(const float* __restrict__ xp,
                  const float* __restrict__ att_src,
                  const float* __restrict__ att_dst,
                  float* __restrict__ ssrc, float* __restrict__ sdst, int Nn)
{
    const int w    = blockIdx.x * 8 + (threadIdx.x >> 5);
    const int lane = threadIdx.x & 31;
    if (w >= 3 * Nn * 4) return;
    const int h = w & 3;
    const int sn = w >> 2;            // s*Nn + n
    const int s  = sn / Nn;

    const float4 xv = *reinterpret_cast<const float4*>(xp + (long)sn * 512 + h * 128 + lane * 4);
    const float4 as = *reinterpret_cast<const float4*>(att_src + (s * 4 + h) * 128 + lane * 4);
    const float4 ad = *reinterpret_cast<const float4*>(att_dst + (s * 4 + h) * 128 + lane * 4);
    float vs = xv.x * as.x + xv.y * as.y + xv.z * as.z + xv.w * as.w;
    float vd = xv.x * ad.x + xv.y * ad.y + xv.z * ad.z + xv.w * ad.w;
    for (int off = 16; off > 0; off >>= 1) {
        vs += __shfl_xor(vs, off, 32);
        vd += __shfl_xor(vd, off, 32);
    }
    if (lane == 0) {
        ssrc[(long)sn * 4 + h] = vs;
        sdst[(long)sn * 4 + h] = vd;
    }
}

// ---------------------------------------------------------------------------
// Edge pass 1: segment max of leaky-relu scores per (dst, head)
// ---------------------------------------------------------------------------
__global__ __launch_bounds__(256)
void edge_max_kernel(const long long* __restrict__ ei, int E, int Nn,
                     const float* __restrict__ ssrc, const float* __restrict__ sdst,
                     unsigned* __restrict__ maxenc)
{
    const long Et  = (long)E + Nn;
    const long idx = (long)blockIdx.x * 256 + threadIdx.x;
    if (idx >= 3 * Et) return;
    const int  s = (int)(idx / Et);
    const long e = idx - (long)s * Et;
    long src, dst;
    if (e < E) { src = ei[e]; dst = ei[E + e]; } else { src = dst = e - E; }

    const float4 a = *reinterpret_cast<const float4*>(ssrc + ((long)s * Nn + src) * 4);
    const float4 b = *reinterpret_cast<const float4*>(sdst + ((long)s * Nn + dst) * 4);
    unsigned* mp = maxenc + ((long)s * Nn + dst) * 4;
    atomicMax(mp + 0, fenc(leaky(a.x + b.x)));
    atomicMax(mp + 1, fenc(leaky(a.y + b.y)));
    atomicMax(mp + 2, fenc(leaky(a.z + b.z)));
    atomicMax(mp + 3, fenc(leaky(a.w + b.w)));
}

// ---------------------------------------------------------------------------
// Edge pass 2: e_exp = exp(score - max); accumulate denominators
// ---------------------------------------------------------------------------
__global__ __launch_bounds__(256)
void edge_exp_kernel(const long long* __restrict__ ei, int E, int Nn,
                     const float* __restrict__ ssrc, const float* __restrict__ sdst,
                     const unsigned* __restrict__ maxenc,
                     float* __restrict__ denom, float* __restrict__ eexp)
{
    const long Et  = (long)E + Nn;
    const long idx = (long)blockIdx.x * 256 + threadIdx.x;
    if (idx >= 3 * Et) return;
    const int  s = (int)(idx / Et);
    const long e = idx - (long)s * Et;
    long src, dst;
    if (e < E) { src = ei[e]; dst = ei[E + e]; } else { src = dst = e - E; }

    const float4 a = *reinterpret_cast<const float4*>(ssrc + ((long)s * Nn + src) * 4);
    const float4 b = *reinterpret_cast<const float4*>(sdst + ((long)s * Nn + dst) * 4);
    const unsigned* mp = maxenc + ((long)s * Nn + dst) * 4;
    float* dp = denom + ((long)s * Nn + dst) * 4;
    float* ep = eexp  + ((long)s * Et + e)   * 4;

    float v0 = __expf(leaky(a.x + b.x) - fdec(mp[0]));
    float v1 = __expf(leaky(a.y + b.y) - fdec(mp[1]));
    float v2 = __expf(leaky(a.z + b.z) - fdec(mp[2]));
    float v3 = __expf(leaky(a.w + b.w) - fdec(mp[3]));
    ep[0] = v0; ep[1] = v1; ep[2] = v2; ep[3] = v3;
    atomicAdd(dp + 0, v0); atomicAdd(dp + 1, v1);
    atomicAdd(dp + 2, v2); atomicAdd(dp + 3, v3);
}

// ---------------------------------------------------------------------------
// Edge pass 3: scatter head-averaged message directly into feat32[N, 3*128]
// out_mean[n,f] = (1/H) * sum_e sum_h alpha[e,h] * xp[src_e, h, f]
// one wave per (scale, edge); lane handles 4 f's across all 4 heads
// ---------------------------------------------------------------------------
__global__ __launch_bounds__(256)
void edge_scatter_kernel(const long long* __restrict__ ei, int E, int Nn,
                         const float* __restrict__ eexp,
                         const float* __restrict__ denom,
                         const float* __restrict__ xp,
                         float* __restrict__ feat32)
{
    const long Et = (long)E + Nn;
    const long w  = (long)blockIdx.x * 8 + (threadIdx.x >> 5);
    const int lane = threadIdx.x & 31;
    if (w >= 3 * Et) return;
    const int  s = (int)(w / Et);
    const long e = w - (long)s * Et;
    long src, dst;
    if (e < E) { src = ei[e]; dst = ei[E + e]; } else { src = dst = e - E; }

    const float4 ex = *reinterpret_cast<const float4*>(eexp  + ((long)s * Et + e)   * 4);
    const float4 dn = *reinterpret_cast<const float4*>(denom + ((long)s * Nn + dst) * 4);
    const float a0 = 0.25f * ex.x / dn.x;
    const float a1 = 0.25f * ex.y / dn.y;
    const float a2 = 0.25f * ex.z / dn.z;
    const float a3 = 0.25f * ex.w / dn.w;

    const float* xr = xp + ((long)s * Nn + src) * 512 + lane * 4;
    const float4 x0 = *reinterpret_cast<const float4*>(xr);
    const float4 x1 = *reinterpret_cast<const float4*>(xr + 128);
    const float4 x2 = *reinterpret_cast<const float4*>(xr + 256);
    const float4 x3 = *reinterpret_cast<const float4*>(xr + 384);

    float r0 = a0 * x0.x + a1 * x1.x + a2 * x2.x + a3 * x3.x;
    float r1 = a0 * x0.y + a1 * x1.y + a2 * x2.y + a3 * x3.y;
    float r2 = a0 * x0.z + a1 * x1.z + a2 * x2.z + a3 * x3.z;
    float r3 = a0 * x0.w + a1 * x1.w + a2 * x2.w + a3 * x3.w;

    float* out = feat32 + (long)dst * 384 + s * 128 + lane * 4;
    atomicAdd(out + 0, r0); atomicAdd(out + 1, r1);
    atomicAdd(out + 2, r2); atomicAdd(out + 3, r3);
}

// add per-scale bias and convert to bf16 for GEMM2 A-matrix
__global__ void feat_bias_bf16(const float* __restrict__ feat32,
                               const float* __restrict__ bias, // [3*128] flat
                               __bf16* __restrict__ featb, long total)
{
    long idx = (long)blockIdx.x * 256 + threadIdx.x;
    if (idx >= total) return;
    int c = (int)(idx % 384);
    featb[idx] = (__bf16)(feat32[idx] + bias[c]);
}

// ---------------------------------------------------------------------------
// Graph mean pool
// ---------------------------------------------------------------------------
__global__ __launch_bounds__(128)
void pool_acc_kernel(const float* __restrict__ combined,
                     const long long* __restrict__ batch,
                     float* __restrict__ sums, float* __restrict__ counts)
{
    const int n = blockIdx.x;
    const int f = threadIdx.x;
    const int b = (int)batch[n];
    atomicAdd(&sums[b * 128 + f], combined[(long)n * 128 + f]);
    if (f == 0) atomicAdd(&counts[b], 1.0f);
}

__global__ __launch_bounds__(128)
void pool_final_kernel(const float* __restrict__ sums,
                       const float* __restrict__ counts,
                       float* __restrict__ out)
{
    const int i = blockIdx.x * 128 + threadIdx.x;
    out[i] = sums[i] / fmaxf(counts[i / 128], 1.0f);
}

// ---------------------------------------------------------------------------
extern "C" void kernel_launch(void* const* d_in, const int* in_sizes, int n_in,
                              void* d_out, int out_size, void* d_ws, size_t ws_size,
                              hipStream_t stream)
{
    const float*     x       = (const float*)d_in[0];
    const long long* ei      = (const long long*)d_in[1];
    const long long* batch   = (const long long*)d_in[2];
    const float*     W       = (const float*)d_in[3];
    const float*     att_src = (const float*)d_in[4];
    const float*     att_dst = (const float*)d_in[5];
    const float*     bias    = (const float*)d_in[6];
    const float*     Wc      = (const float*)d_in[7];
    const float*     bc      = (const float*)d_in[8];
    float* out = (float*)d_out;
    (void)n_in; (void)out_size; (void)ws_size;

    const int Din = 256, H = 4, F = 128, S = 3, HF = 512, SF = 384, G = 64;
    const int N = in_sizes[0] / Din;          // 20000
    const int E = in_sizes[1] / 2;            // 320000
    const long Et = (long)E + N;              // + self loops

    char* ws = (char*)d_ws;
    size_t off = 0;
    auto alloc = [&](size_t bytes) -> void* {
        off = (off + 255) & ~(size_t)255;
        void* p = ws + off;
        off += bytes;
        return p;
    };

    // ---- zero-initialized region (contiguous prefix of ws) ----
    unsigned* maxenc = (unsigned*)alloc((size_t)S * N * H * 4);
    float*    denom  = (float*)   alloc((size_t)S * N * H * 4);
    float*    feat32 = (float*)   alloc((size_t)N * SF * 4);
    float*    sums   = (float*)   alloc((size_t)G * F * 4);
    float*    counts = (float*)   alloc((size_t)G * 4);
    const long zeroFloats = (long)(off / 4);

    // ---- remaining scratch ----
    __bf16* xb       = (__bf16*)alloc((size_t)N * Din * 2);
    __bf16* Wt       = (__bf16*)alloc((size_t)S * HF * Din * 2);
    __bf16* Wct      = (__bf16*)alloc((size_t)F * SF * 2);
    float*  xp       = (float*) alloc((size_t)S * N * HF * 4);
    float*  ssrc     = (float*) alloc((size_t)S * N * H * 4);
    float*  sdst     = (float*) alloc((size_t)S * N * H * 4);
    float*  eexp     = (float*) alloc((size_t)S * Et * H * 4);
    __bf16* featb    = (__bf16*)alloc((size_t)N * SF * 2);
    float*  combined = (float*) alloc((size_t)N * F * 4);

    const dim3 b256(256);

    // 1) zero atomic accumulators
    zero_kernel<<<(unsigned)((zeroFloats + 255) / 256), b256, 0, stream>>>((float*)ws, zeroFloats);

    // 2) precision conversions / weight transposes
    const long nx = (long)N * Din;
    f2bf_kernel<<<(unsigned)((nx + 255) / 256), b256, 0, stream>>>(x, xb, nx);
    const long nw = (long)S * Din * HF;
    transpose_f2bf<<<(unsigned)((nw + 255) / 256), b256, 0, stream>>>(W, Wt, Din, HF, nw);
    const long nwc = (long)SF * F;
    transpose_f2bf<<<(unsigned)((nwc + 255) / 256), b256, 0, stream>>>(Wc, Wct, SF, F, nwc);

    // 3) xp[s] = x @ W[s]   (bf16 WMMA, f32 accumulate)  [20000,256]x[256,512] x3
    dim3 g1(N / 16, HF / 64, S);
    gemm_bf16_wmma<256><<<g1, dim3(32), 0, stream>>>(xb, 0L, Wt, (long)HF * Din,
                                                     xp, (long)N * HF, nullptr, HF);

    // 4) attention projections
    const int nwav = S * N * H;
    sdots_kernel<<<(unsigned)((nwav + 7) / 8), b256, 0, stream>>>(xp, att_src, att_dst, ssrc, sdst, N);

    // 5) segment softmax over incoming edges + message scatter
    const long nt = 3 * Et;
    edge_max_kernel<<<(unsigned)((nt + 255) / 256), b256, 0, stream>>>(ei, E, N, ssrc, sdst, maxenc);
    edge_exp_kernel<<<(unsigned)((nt + 255) / 256), b256, 0, stream>>>(ei, E, N, ssrc, sdst, maxenc, denom, eexp);
    edge_scatter_kernel<<<(unsigned)((nt + 7) / 8), b256, 0, stream>>>(ei, E, N, eexp, denom, xp, feat32);

    // 6) bias + bf16 cast for combiner GEMM
    const long nf = (long)N * SF;
    feat_bias_bf16<<<(unsigned)((nf + 255) / 256), b256, 0, stream>>>(feat32, bias, featb, nf);

    // 7) combined = concat3 @ Wc + bc   [20000,384]x[384,128]  (WMMA)
    dim3 g2(N / 16, F / 64, 1);
    gemm_bf16_wmma<384><<<g2, dim3(32), 0, stream>>>(featb, 0L, Wct, 0L,
                                                     combined, 0L, bc, F);

    // 8) global mean pool per graph
    pool_acc_kernel<<<N, dim3(128), 0, stream>>>(combined, batch, sums, counts);
    pool_final_kernel<<<G, dim3(128), 0, stream>>>(sums, counts, out);
}
